// DataExpander_42769284334217
// MI455X (gfx1250) — matrix operand order repaired and checked by reading, not compiled, not run on hardware
//
#include <hip/hip_runtime.h>
#include <hip/hip_bf16.h>
#include <math.h>

// ---------------------------------------------------------------------------
// DataExpander pipeline for MI455X (gfx1250, wave32, WMMA).
//   GEMM1: x[512,20000]       @ W_ge.T -> combined[:, 0:1024]   (+bias,lrelu)
//   GEMM2: emb[20000,512]     @ W_em.T -> geneT bf16 [1024,20000] (+bias,lrelu)
//   SMAX : softmax rows of x -> weights fp32 [512,20000]
//   GEMM3: weights            @ geneT.T -> combined[:,1024:2048]
//   GEMM4: combined[512,2048] @ W_c.T  -> out[512,2048]         (+bias,lrelu)
// All GEMMs use v_wmma_f32_16x16x32_bf16 with in-register fp32->bf16 convert.
// K=20000 GEMMs use a 64x32 wave tile (MT=4) to halve L2 traffic on the
// dominant W_ge / geneT streams; other GEMMs use 32x32 (MT=2).
// ---------------------------------------------------------------------------

typedef __attribute__((ext_vector_type(8)))  float          v8f;
typedef __attribute__((ext_vector_type(16))) __bf16         v16bf;
typedef __attribute__((ext_vector_type(16))) unsigned short v16us;

#define BATCH   512
#define INPUT   20000
#define HIDDEN  1024
#define EMBED   512
#define OUTPUT  2048

// ---- fragment builders (contiguous K per lane half, same order for A and B) ----------------------

// A-matrix 16x32 bf16 fragment. p = &A[row][k0]; contiguous in K.
// lanes 0-15: K=0..15 ; lanes 16-31: K=16..31 ; VGPR v holds K = half*16+2v,+1
__device__ __forceinline__ v16bf frag_a_f32(const float* __restrict__ p, int half) {
    v16bf f;
#pragma unroll
    for (int v = 0; v < 8; ++v) {
        int kb = half * 16 + (v << 1);
        float2 t = *reinterpret_cast<const float2*>(p + kb);
        f[2 * v]     = (__bf16)t.x;
        f[2 * v + 1] = (__bf16)t.y;
    }
    return f;
}

// B-matrix 32x16 bf16 fragment from fp32 memory stored [N][K] (K-contiguous).
// lanes 0-15: K=0..15 ; lanes 16-31: K=16..31 ; VGPR v holds K = half*16+2v,+1
__device__ __forceinline__ v16bf frag_b_f32(const float* __restrict__ p, int half) {
    v16bf f;
#pragma unroll
    for (int v = 0; v < 8; ++v) {
        int kb = half * 16 + (v << 1);
        float2 t = *reinterpret_cast<const float2*>(p + kb);
        f[2 * v]     = (__bf16)t.x;
        f[2 * v + 1] = (__bf16)t.y;
    }
    return f;
}

// Same, but B already packed bf16 in memory ([N][K], K-contiguous).
__device__ __forceinline__ v16bf frag_b_bf16(const unsigned short* __restrict__ p, int half) {
    v16us u;
#pragma unroll
    for (int v = 0; v < 8; ++v) {
        int kb = half * 16 + (v << 1);
        unsigned int w = *reinterpret_cast<const unsigned int*>(p + kb);
        u[2 * v]     = (unsigned short)(w & 0xFFFFu);
        u[2 * v + 1] = (unsigned short)(w >> 16);
    }
    return __builtin_bit_cast(v16bf, u);
}

__device__ __forceinline__ v8f wmma_bf16(v16bf a, v16bf b, v8f c) {
    return __builtin_amdgcn_wmma_f32_16x16x32_bf16(
        /*neg_a=*/false, a, /*neg_b=*/false, b,
        /*c_mod=*/(short)0, c, /*reuse_a=*/false, /*reuse_b=*/false);
}

// ---------------------------------------------------------------------------
// Generic wave-tile GEMM: each wave produces a (MT*16)x32 tile of
// D = act(A*B^T + bias).
//   A: fp32 [M][lda]           (K-contiguous)
//   B: [N][ldb], fp32 or bf16  (K-contiguous; this is W row-major == B^T)
//   D: if STORE_T: bf16 [N][ldd] (transposed store), else fp32 [M][ldd] at +dcol
// Requires M%(16*MT)==0, N%32==0, K%32==0,
//          grid*8 waves == (M/(16*MT))*(N/32).
// ---------------------------------------------------------------------------
template <int MT, bool B_IS_BF16, bool BIAS_LRELU, bool STORE_T>
__global__ void wmma_gemm_kernel(const float* __restrict__ A,
                                 const void*  __restrict__ Bv,
                                 const float* __restrict__ bias,
                                 void* __restrict__ Dv,
                                 int M, int N, int K,
                                 int lda, int ldb, int ldd, int dcol) {
    const int wave = (blockIdx.x * blockDim.x + threadIdx.x) >> 5;
    const int lane = threadIdx.x & 31;
    const int half = lane >> 4;
    const int r    = lane & 15;

    const int ntiles = N >> 5;              // 32-wide N macro-tiles
    const int tm = wave / ntiles;
    const int tn = wave % ntiles;
    if (tm * (16 * MT) >= M) return;        // wave-uniform

    const int m0 = tm * (16 * MT);
    const int n0 = tn << 5;

    v8f acc[MT][2];
#pragma unroll
    for (int mt = 0; mt < MT; ++mt) {
        acc[mt][0] = v8f{};
        acc[mt][1] = v8f{};
    }

    const float* ap[MT];
#pragma unroll
    for (int mt = 0; mt < MT; ++mt)
        ap[mt] = A + (size_t)(m0 + mt * 16 + r) * lda;

    for (int k0 = 0; k0 < K; k0 += 32) {
        v16bf b0, b1;
        if (B_IS_BF16) {
            const unsigned short* B = (const unsigned short*)Bv;
            b0 = frag_b_bf16(B + (size_t)(n0 + r) * ldb + k0, half);
            b1 = frag_b_bf16(B + (size_t)(n0 + 16 + r) * ldb + k0, half);
        } else {
            const float* B = (const float*)Bv;
            b0 = frag_b_f32(B + (size_t)(n0 + r) * ldb + k0, half);
            b1 = frag_b_f32(B + (size_t)(n0 + 16 + r) * ldb + k0, half);
        }
#pragma unroll
        for (int mt = 0; mt < MT; ++mt) {
            v16bf a = frag_a_f32(ap[mt] + k0, half);
            acc[mt][0] = wmma_bf16(a, b0, acc[mt][0]);
            acc[mt][1] = wmma_bf16(a, b1, acc[mt][1]);
        }
    }

    // Epilogue: C/D layout -> VGPR rr: M = rr + half*8, N = lane%16.
#pragma unroll
    for (int mt = 0; mt < MT; ++mt) {
#pragma unroll
        for (int nt = 0; nt < 2; ++nt) {
            const int nglob = n0 + nt * 16 + r;
            const float bv = BIAS_LRELU ? bias[nglob] : 0.0f;
#pragma unroll
            for (int rr = 0; rr < 8; ++rr) {
                const int mglob = m0 + mt * 16 + rr + half * 8;
                float val = acc[mt][nt][rr] + bv;
                if (BIAS_LRELU) val = (val > 0.0f) ? val : 0.01f * val;
                if (STORE_T) {
                    ((__bf16*)Dv)[(size_t)nglob * ldd + mglob] = (__bf16)val;
                } else {
                    ((float*)Dv)[(size_t)mglob * ldd + dcol + nglob] = val;
                }
            }
        }
    }
}

// ---------------------------------------------------------------------------
// Row softmax: one 256-thread block per row (8 waves, wave32).
// ---------------------------------------------------------------------------
__global__ void softmax_rows_kernel(const float* __restrict__ X,
                                    float* __restrict__ W, int cols) {
    const int row = blockIdx.x;
    const float* x = X + (size_t)row * cols;
    float* w = W + (size_t)row * cols;

    __shared__ float sred[8];
    const int tid = threadIdx.x;
    const int lane = tid & 31;
    const int wv = tid >> 5;

    // ---- pass 1: row max ----
    float lmax = -INFINITY;
    for (int c = tid; c < cols; c += blockDim.x) lmax = fmaxf(lmax, x[c]);
#pragma unroll
    for (int o = 16; o > 0; o >>= 1) lmax = fmaxf(lmax, __shfl_xor(lmax, o, 32));
    if (lane == 0) sred[wv] = lmax;
    __syncthreads();
    float rmax = sred[0];
#pragma unroll
    for (int i = 1; i < 8; ++i) rmax = fmaxf(rmax, sred[i]);
    __syncthreads();

    // ---- pass 2: sum of exp ----
    float lsum = 0.0f;
    for (int c = tid; c < cols; c += blockDim.x) lsum += __expf(x[c] - rmax);
#pragma unroll
    for (int o = 16; o > 0; o >>= 1) lsum += __shfl_xor(lsum, o, 32);
    if (lane == 0) sred[wv] = lsum;
    __syncthreads();
    float rsum = 0.0f;
#pragma unroll
    for (int i = 0; i < 8; ++i) rsum += sred[i];
    const float inv = 1.0f / rsum;

    // ---- pass 3: write normalized weights ----
    for (int c = tid; c < cols; c += blockDim.x)
        w[c] = __expf(x[c] - rmax) * inv;
}

// ---------------------------------------------------------------------------

extern "C" void kernel_launch(void* const* d_in, const int* in_sizes, int n_in,
                              void* d_out, int out_size, void* d_ws, size_t ws_size,
                              hipStream_t stream) {
    (void)in_sizes; (void)n_in; (void)out_size; (void)ws_size;

    const float* x     = (const float*)d_in[0];   // [512, 20000]
    const float* emb   = (const float*)d_in[1];   // [20000, 512]
    const float* W_ge  = (const float*)d_in[2];   // [1024, 20000]
    const float* b_ge  = (const float*)d_in[3];   // [1024]
    const float* W_em  = (const float*)d_in[4];   // [1024, 512]
    const float* b_em  = (const float*)d_in[5];   // [1024]
    const float* W_c   = (const float*)d_in[6];   // [2048, 2048]
    const float* b_c   = (const float*)d_in[7];   // [2048]
    float* out = (float*)d_out;                   // [512, 2048]

    // Workspace layout
    char* ws = (char*)d_ws;
    const size_t weights_bytes = (size_t)BATCH * INPUT * sizeof(float);     // 40.96 MB
    const size_t geneT_bytes   = (size_t)HIDDEN * INPUT * sizeof(__bf16);   // 40.96 MB
    float*  weights  = (float*)ws;                                    // [512, 20000] fp32
    __bf16* geneT    = (__bf16*)(ws + weights_bytes);                 // [1024, 20000] bf16
    float*  combined = (float*)(ws + weights_bytes + geneT_bytes);    // [512, 2048] fp32

    // 1) softmax(x) -> weights
    softmax_rows_kernel<<<BATCH, 256, 0, stream>>>(x, weights, INPUT);

    // 2) GEMM1: x @ W_ge.T (+b_ge, lrelu) -> combined[:, 0:1024]   (64x32 tiles)
    {
        int waves = (BATCH / 64) * (HIDDEN / 32);        // 256
        wmma_gemm_kernel<4, false, true, false><<<waves / 8, 256, 0, stream>>>(
            x, W_ge, b_ge, combined,
            BATCH, HIDDEN, INPUT, INPUT, INPUT, 2 * HIDDEN, 0);
    }

    // 3) GEMM2: emb @ W_em.T (+b_em, lrelu) -> geneT[hidden][gene] bf16 (transposed)
    {
        int waves = (INPUT / 32) * (HIDDEN / 32);        // 20000
        wmma_gemm_kernel<2, false, true, true><<<waves / 8, 256, 0, stream>>>(
            emb, W_em, b_em, geneT,
            INPUT, HIDDEN, EMBED, EMBED, EMBED, INPUT, 0);
    }

    // 4) GEMM3: weights @ geneT.T -> combined[:, 1024:2048]  (64x32 tiles, no bias/act)
    {
        int waves = (BATCH / 64) * (HIDDEN / 32);        // 256
        wmma_gemm_kernel<4, true, false, false><<<waves / 8, 256, 0, stream>>>(
            weights, geneT, nullptr, combined,
            BATCH, HIDDEN, INPUT, INPUT, INPUT, 2 * HIDDEN, HIDDEN);
    }

    // 5) GEMM4: combined @ W_c.T (+b_c, lrelu) -> out
    {
        int waves = (BATCH / 32) * (OUTPUT / 32);        // 1024
        wmma_gemm_kernel<2, false, true, false><<<waves / 8, 256, 0, stream>>>(
            combined, W_c, b_c, out,
            BATCH, OUTPUT, 2 * HIDDEN, 2 * HIDDEN, 2 * HIDDEN, OUTPUT, 0);
    }
}
